// AttnPretrainGNN_5488968204771
// MI455X (gfx1250) — matrix operand order repaired and checked by reading, not compiled, not run on hardware
//
#include <hip/hip_runtime.h>

// ---------------------------------------------------------------------------
// Hetero 2-layer GAT + edge decoders for MI455X (gfx1250, wave32).
// Dense transforms use v_wmma_f32_16x16x32_f16 (f16 in, f32 accumulate).
// Weights pre-swizzled into per-lane fragment order (one 32B load per lane
// per k-step); activations pre-converted to f16 (two 16B loads per k-step).
// Out-of-range rows are handled by clamping the A row pointer (C row m only
// depends on A row m), so the fully-unrolled inner loop is unconditional:
// 4x global_load_b128 + 1x v_wmma.  Edge softmax/scatter passes are
// HBM-bound f32 with hardware float atomics.
// ---------------------------------------------------------------------------

#define NG   100000
#define NDD  50000
#define FG   128
#define FD   64
#define HDIM 128
#define NE   500000
#define NL   200000

typedef __attribute__((ext_vector_type(16))) _Float16 v16h;
typedef __attribute__((ext_vector_type(8)))  _Float16 v8h;
typedef __attribute__((ext_vector_type(8)))  float    v8f;

// ------------------------- helpers -----------------------------------------

__device__ __forceinline__ unsigned f32_ord(float f) {
    unsigned u = __float_as_uint(f);
    return (u & 0x80000000u) ? ~u : (u | 0x80000000u);
}
__device__ __forceinline__ float ord_f32(unsigned u) {
    return (u & 0x80000000u) ? __uint_as_float(u & 0x7fffffffu)
                             : __uint_as_float(~u);
}
__device__ __forceinline__ void atomAddF(float* p, float v) {
    unsafeAtomicAdd(p, v);   // global_atomic_add_f32
}

__global__ void fill_u32_k(unsigned* __restrict__ p, unsigned v, long n) {
    long i = (long)blockIdx.x * blockDim.x + threadIdx.x;
    if (i < n) p[i] = v;
}

__global__ void cvt_f16_k(const float* __restrict__ s, _Float16* __restrict__ d, long n) {
    long i = (long)blockIdx.x * blockDim.x + threadIdx.x;
    if (i < n) d[i] = (_Float16)s[i];
}

// Pre-swizzle a [K,128] f32 weight into WMMA B-fragment order, f16:
//   Bs[((kb/32)*2 + g)*128 + n][j] = (f16) W[kb + 8g + j + (j>=8 ? 8 : 0)][n],  j = 0..15
// One thread per (kb_idx, g, n); each writes 16 contiguous halves (32B).
__global__ void wswz_k(const float* __restrict__ W, _Float16* __restrict__ Bs, int K) {
    int t = blockIdx.x * blockDim.x + threadIdx.x;
    int total = (K >> 5) * 2 * 128;
    if (t >= total) return;
    int n   = t & 127;
    int g   = (t >> 7) & 1;
    int kbi = t >> 8;
    const float* src = W + (long)(kbi * 32 + 8 * g) * 128 + n;
    _Float16* dst = Bs + (long)t * 16;
#pragma unroll
    for (int j = 0; j < 16; ++j) dst[j] = (_Float16)src[(j < 8 ? j : j + 8) * 128];
}

// ------------------------- WMMA GEMM ----------------------------------------
// C[M,128] = act(A16[M,K] @ W + bias), W pre-swizzled f16.  K in {64,128,256}.
// Block = 256 threads = 8 waves; wave w -> cols [16w,16w+16); block -> 16 rows.
// A fragment (16x32 f16, ISA 7.12.2): lane (g=lane>>4, m=lane&15):
//   halves 0..7  = A[m][kb + 8g + j],  halves 8..15 = A[m][kb + 16 + 8g + j]
//   -> two contiguous 16B loads from the f16 row (row index clamped to 0 when
//      out of range; the corresponding C rows are masked at the store).
// B fragment: one contiguous 32B load from the swizzled weight buffer.
// C/D (16x16 f32): vgpr v -> row row0 + v + 8g, col n0 + (lane&15)

__device__ __forceinline__ v16h load_a_frag(const _Float16* Ap, int koff) {
    v8h a0 = *(const v8h*)(Ap + koff);
    v8h a1 = *(const v8h*)(Ap + koff + 16);
    v16h a;
#pragma unroll
    for (int j = 0; j < 8; ++j) { a[j] = a0[j]; a[j + 8] = a1[j]; }
    return a;
}

template <int K>
__global__ __launch_bounds__(256)
void wmma_gemm_k(const _Float16* __restrict__ A, const _Float16* __restrict__ Bs,
                 const float* __restrict__ bias, float* __restrict__ C,
                 int M, int relu)
{
    const int wave = threadIdx.x >> 5;
    const int lane = threadIdx.x & 31;
    const int g    = lane >> 4;
    const int mn   = lane & 15;
    const long row0 = (long)blockIdx.x << 4;
    const int  n0   = wave << 4;
    const long arow = row0 + mn;
    const long arc  = (arow < (long)M) ? arow : 0;        // clamp: safe memory
    const _Float16* Ap = A + arc * (long)K + 8 * g;
    const _Float16* Bp = Bs + ((long)g * 128 + n0 + mn) * 16;

    v8f acc = {0.f,0.f,0.f,0.f,0.f,0.f,0.f,0.f};
#pragma unroll
    for (int kb = 0; kb < K; kb += 32) {
        v16h a = load_a_frag(Ap, kb);
        v16h b = *(const v16h*)(Bp + (long)(kb >> 5) * 2 * 128 * 16);
        acc = __builtin_amdgcn_wmma_f32_16x16x32_f16(false, a, false, b,
                                                     (short)0, acc, false, false);
    }
    const int col = n0 + mn;
    const float bv = bias ? bias[col] : 0.0f;
#pragma unroll
    for (int v = 0; v < 8; ++v) {
        long r = row0 + v + 8 * g;
        if (r < (long)M) {
            float x = acc[v] + bv;
            if (relu) x = fmaxf(x, 0.0f);
            C[r * 128 + col] = x;
        }
    }
}

// Decoder GEMM with fused gather: row r of A is concat(Zs16[isrc[r]], Zd16[idst[r]]),
// K fixed 256.  kb<128 reads the Zs half, kb>=128 the Zd half.
__global__ __launch_bounds__(256)
void wmma_gemm_pair_k(const _Float16* __restrict__ Zs, const _Float16* __restrict__ Zd,
                      const int* __restrict__ isrc, const int* __restrict__ idst,
                      const _Float16* __restrict__ Bs, const float* __restrict__ bias,
                      float* __restrict__ C, int M, int relu)
{
    const int wave = threadIdx.x >> 5;
    const int lane = threadIdx.x & 31;
    const int g    = lane >> 4;
    const int mn   = lane & 15;
    const long row0 = (long)blockIdx.x << 4;
    const int  n0   = wave << 4;
    const long arow = row0 + mn;
    const bool rv   = arow < (long)M;
    const int i0 = rv ? isrc[arow] : 0;                   // clamp: safe memory
    const int i1 = rv ? idst[arow] : 0;
    const _Float16* As = Zs + (long)i0 * 128 + 8 * g;
    const _Float16* Ad = Zd + (long)i1 * 128 + 8 * g;
    const _Float16* Bp = Bs + ((long)g * 128 + n0 + mn) * 16;

    v8f acc = {0.f,0.f,0.f,0.f,0.f,0.f,0.f,0.f};
#pragma unroll
    for (int kb = 0; kb < 256; kb += 32) {
        const _Float16* Ap = (kb < 128) ? (As + kb) : (Ad + (kb - 128));
        v16h a = load_a_frag(Ap, 0);
        v16h b = *(const v16h*)(Bp + (long)(kb >> 5) * 2 * 128 * 16);
        acc = __builtin_amdgcn_wmma_f32_16x16x32_f16(false, a, false, b,
                                                     (short)0, acc, false, false);
    }
    const int col = n0 + mn;
    const float bv = bias ? bias[col] : 0.0f;
#pragma unroll
    for (int v = 0; v < 8; ++v) {
        long r = row0 + v + 8 * g;
        if (r < (long)M) {
            float x = acc[v] + bv;
            if (relu) x = fmaxf(x, 0.0f);
            C[r * 128 + col] = x;
        }
    }
}

// ------------------------- row-wise 128-dot (wave per row) ------------------
__global__ void rowdot_k(const float* __restrict__ X, const float* __restrict__ w,
                         const float* __restrict__ bsc, float* __restrict__ out, int N)
{
    int row  = blockIdx.x * (blockDim.x >> 5) + (threadIdx.x >> 5);
    int lane = threadIdx.x & 31;
    if (row >= N) return;
    float4 x  = ((const float4*)(X + (long)row * 128))[lane];
    float4 wv = ((const float4*)w)[lane];
    float s = x.x*wv.x + x.y*wv.y + x.z*wv.z + x.w*wv.w;
#pragma unroll
    for (int off = 16; off; off >>= 1) s += __shfl_down(s, off);
    if (lane == 0) out[row] = s + (bsc ? bsc[0] : 0.0f);
}

// ------------------------- edge softmax passes ------------------------------

__global__ void edge1_k(const int* __restrict__ ei, int E,
                        const float* __restrict__ as_, const float* __restrict__ ad_,
                        float* __restrict__ score, unsigned* __restrict__ mbuf)
{
    int e = blockIdx.x * blockDim.x + threadIdx.x;
    if (e >= E) return;
    int r = ei[e], c = ei[E + e];
    float sc = as_[r] + ad_[c];
    sc = (sc > 0.0f) ? sc : 0.2f * sc;          // leaky_relu(0.2)
    score[e] = sc;
    atomicMax(&mbuf[c], f32_ord(sc));           // order-preserving uint map
}

__global__ void edge2_k(const int* __restrict__ ei, int E,
                        const float* __restrict__ score, const unsigned* __restrict__ mbuf,
                        float* __restrict__ eexp, float* __restrict__ ssum)
{
    int e = blockIdx.x * blockDim.x + threadIdx.x;
    if (e >= E) return;
    int c = ei[E + e];
    float ex = __expf(score[e] - ord_f32(mbuf[c]));
    eexp[e] = ex;
    atomAddF(&ssum[c], ex);
}

// wave per edge: out[col] += att * hs[row]  (128 dims, float4 per lane)
__global__ void edge3_k(const int* __restrict__ ei, int E,
                        const float* __restrict__ eexp, const float* __restrict__ ssum,
                        const float* __restrict__ hs, float* __restrict__ out)
{
    long e  = (long)blockIdx.x * (blockDim.x >> 5) + (threadIdx.x >> 5);
    int lane = threadIdx.x & 31;
    if (e >= E) return;
    int r = ei[e], c = ei[E + e];
    float w = eexp[e] / (ssum[c] + 1e-16f);
    float4 h = ((const float4*)(hs + (long)r * 128))[lane];
    float* op = out + (long)c * 128 + lane * 4;
    atomAddF(op + 0, w * h.x);
    atomAddF(op + 1, w * h.y);
    atomAddF(op + 2, w * h.z);
    atomAddF(op + 3, w * h.w);
}

__global__ void bias_act_k(float* __restrict__ X, const float* __restrict__ b,
                           long n, int relu)
{
    long i = (long)blockIdx.x * blockDim.x + threadIdx.x;
    if (i >= n) return;
    float v = X[i] + b[i & 127];
    X[i] = relu ? fmaxf(v, 0.0f) : v;
}

// ------------------------- host orchestration -------------------------------

static inline void launch_gemm(const _Float16* A, const _Float16* Bs, const float* bias,
                               float* C, int M, int K, int relu, hipStream_t stream)
{
    unsigned grid = (unsigned)((M + 15) / 16);
    if (K == 64)       wmma_gemm_k<64><<<grid, 256, 0, stream>>>(A, Bs, bias, C, M, relu);
    else if (K == 128) wmma_gemm_k<128><<<grid, 256, 0, stream>>>(A, Bs, bias, C, M, relu);
    else               wmma_gemm_k<256><<<grid, 256, 0, stream>>>(A, Bs, bias, C, M, relu);
}

extern "C" void kernel_launch(void* const* d_in, const int* in_sizes, int n_in,
                              void* d_out, int out_size, void* d_ws, size_t ws_size,
                              hipStream_t stream)
{
    (void)in_sizes; (void)n_in; (void)out_size; (void)ws_size;

    const float* x_gene = (const float*)d_in[0];
    const float* x_dis  = (const float*)d_in[1];
    const int* ei_gd  = (const int*)d_in[2];
    const int* ei_dg  = (const int*)d_in[3];
    const int* eli_gd = (const int*)d_in[4];
    const int* eli_dg = (const int*)d_in[5];

    // params flattened depth-first in dict insertion order
    const float* c1gd_Ws = (const float*)d_in[6];   // [128,128]
    const float* c1gd_Wd = (const float*)d_in[7];   // [64,128]
    const float* c1gd_as = (const float*)d_in[8];
    const float* c1gd_ad = (const float*)d_in[9];
    const float* c1gd_b  = (const float*)d_in[10];
    const float* c1dg_Ws = (const float*)d_in[11];  // [64,128]
    const float* c1dg_Wd = (const float*)d_in[12];  // [128,128]
    const float* c1dg_as = (const float*)d_in[13];
    const float* c1dg_ad = (const float*)d_in[14];
    const float* c1dg_b  = (const float*)d_in[15];
    const float* c2gd_Ws = (const float*)d_in[16];
    const float* c2gd_Wd = (const float*)d_in[17];
    const float* c2gd_as = (const float*)d_in[18];
    const float* c2gd_ad = (const float*)d_in[19];
    const float* c2gd_b  = (const float*)d_in[20];
    const float* c2dg_Ws = (const float*)d_in[21];
    const float* c2dg_Wd = (const float*)d_in[22];
    const float* c2dg_as = (const float*)d_in[23];
    const float* c2dg_ad = (const float*)d_in[24];
    const float* c2dg_b  = (const float*)d_in[25];
    const float* dgd_W1  = (const float*)d_in[26];  // [256,128]
    const float* dgd_b1  = (const float*)d_in[27];
    const float* dgd_W2  = (const float*)d_in[28];  // [128,1]
    const float* dgd_b2  = (const float*)d_in[29];  // scalar
    const float* ddg_W1  = (const float*)d_in[30];
    const float* ddg_b1  = (const float*)d_in[31];
    const float* ddg_W2  = (const float*)d_in[32];
    const float* ddg_b2  = (const float*)d_in[33];

    // outputs
    float* z_gene = (float*)d_out;
    float* z_dis  = z_gene + (size_t)NG * 128;
    float* s_gd   = z_dis  + (size_t)NDD * 128;
    float* s_dg   = s_gd + NL;

    // workspace carve-out
    size_t off = 0;
    auto alloc = [&](size_t bytes) -> void* {
        void* p = (char*)d_ws + off;
        off = (off + bytes + 255) & ~(size_t)255;
        return p;
    };
    _Float16* w16_c1gd_s = (_Float16*)alloc(16384 * 2);
    _Float16* w16_c1gd_d = (_Float16*)alloc(8192 * 2);
    _Float16* w16_c1dg_s = (_Float16*)alloc(8192 * 2);
    _Float16* w16_c1dg_d = (_Float16*)alloc(16384 * 2);
    _Float16* w16_c2gd_s = (_Float16*)alloc(16384 * 2);
    _Float16* w16_c2gd_d = (_Float16*)alloc(16384 * 2);
    _Float16* w16_c2dg_s = (_Float16*)alloc(16384 * 2);
    _Float16* w16_c2dg_d = (_Float16*)alloc(16384 * 2);
    _Float16* w16_dgd    = (_Float16*)alloc(32768 * 2);
    _Float16* w16_ddg    = (_Float16*)alloc(32768 * 2);
    float* hs_buf  = (float*)alloc((size_t)NG * 128 * 4);
    float* hd_buf  = (float*)alloc((size_t)NG * 128 * 4);
    float* h_gene  = (float*)alloc((size_t)NG * 128 * 4);
    float* h_dis   = (float*)alloc((size_t)NDD * 128 * 4);
    float* alpha_s = (float*)alloc((size_t)NG * 4);
    float* alpha_d = (float*)alloc((size_t)NG * 4);
    float* score   = (float*)alloc((size_t)NE * 4);
    float* eexp    = (float*)alloc((size_t)NE * 4);
    unsigned* mbuf = (unsigned*)alloc((size_t)NG * 4);
    float* ssum    = (float*)alloc((size_t)NG * 4);
    float* Hdec    = (float*)alloc((size_t)NL * 128 * 4);
    // f16 activation copies
    _Float16* xg16 = (_Float16*)alloc((size_t)NG * FG * 2);
    _Float16* xd16 = (_Float16*)alloc((size_t)NDD * FD * 2);
    _Float16* hg16 = (_Float16*)alloc((size_t)NG * 128 * 2);
    _Float16* hd16 = (_Float16*)alloc((size_t)NDD * 128 * 2);
    _Float16* zg16 = (_Float16*)alloc((size_t)NG * 128 * 2);
    _Float16* zd16 = (_Float16*)alloc((size_t)NDD * 128 * 2);

    auto swz = [&](const float* w, _Float16* dst, int K) {
        int total = (K >> 5) * 2 * 128;
        wswz_k<<<(total + 255) / 256, 256, 0, stream>>>(w, dst, K);
    };
    swz(c1gd_Ws, w16_c1gd_s, 128); swz(c1gd_Wd, w16_c1gd_d, 64);
    swz(c1dg_Ws, w16_c1dg_s, 64);  swz(c1dg_Wd, w16_c1dg_d, 128);
    swz(c2gd_Ws, w16_c2gd_s, 128); swz(c2gd_Wd, w16_c2gd_d, 128);
    swz(c2dg_Ws, w16_c2dg_s, 128); swz(c2dg_Wd, w16_c2dg_d, 128);
    swz(dgd_W1,  w16_dgd,    256); swz(ddg_W1,  w16_ddg,    256);

    auto cvt = [&](const float* s, _Float16* d, long n) {
        cvt_f16_k<<<(unsigned)((n + 255) / 256), 256, 0, stream>>>(s, d, n);
    };
    cvt(x_gene, xg16, (long)NG * FG);
    cvt(x_dis,  xd16, (long)NDD * FD);

    auto run_gat = [&](const _Float16* Xs, int Ns, int Ks,
                       const _Float16* Xd, int Nd, int Kd,
                       const _Float16* Ws16, const _Float16* Wd16,
                       const float* avs, const float* avd, const float* bvec,
                       const int* ei, float* out, int relu) {
        launch_gemm(Xs, Ws16, nullptr, hs_buf, Ns, Ks, 0, stream);
        launch_gemm(Xd, Wd16, nullptr, hd_buf, Nd, Kd, 0, stream);
        rowdot_k<<<(Ns + 7) / 8, 256, 0, stream>>>(hs_buf, avs, nullptr, alpha_s, Ns);
        rowdot_k<<<(Nd + 7) / 8, 256, 0, stream>>>(hd_buf, avd, nullptr, alpha_d, Nd);
        fill_u32_k<<<(Nd + 255) / 256, 256, 0, stream>>>(mbuf, 0u, Nd);
        fill_u32_k<<<(Nd + 255) / 256, 256, 0, stream>>>((unsigned*)ssum, 0u, Nd);
        long nout = (long)Nd * 128;
        fill_u32_k<<<(unsigned)((nout + 255) / 256), 256, 0, stream>>>((unsigned*)out, 0u, nout);
        edge1_k<<<(NE + 255) / 256, 256, 0, stream>>>(ei, NE, alpha_s, alpha_d, score, mbuf);
        edge2_k<<<(NE + 255) / 256, 256, 0, stream>>>(ei, NE, score, mbuf, eexp, ssum);
        edge3_k<<<(NE + 7) / 8, 256, 0, stream>>>(ei, NE, eexp, ssum, hs_buf, out);
        bias_act_k<<<(unsigned)((nout + 255) / 256), 256, 0, stream>>>(out, bvec, nout, relu);
    };

    // Layer 1 (ReLU): gene->dis and dis->gene
    run_gat(xg16, NG, FG, xd16, NDD, FD, w16_c1gd_s, w16_c1gd_d,
            c1gd_as, c1gd_ad, c1gd_b, ei_gd, h_dis, 1);
    run_gat(xd16, NDD, FD, xg16, NG, FG, w16_c1dg_s, w16_c1dg_d,
            c1dg_as, c1dg_ad, c1dg_b, ei_dg, h_gene, 1);

    // f16 copies of layer-1 outputs for the layer-2 GEMMs
    cvt(h_gene, hg16, (long)NG * 128);
    cvt(h_dis,  hd16, (long)NDD * 128);

    // Layer 2 (linear): writes directly into d_out z regions
    run_gat(hg16, NG, HDIM, hd16, NDD, HDIM, w16_c2gd_s, w16_c2gd_d,
            c2gd_as, c2gd_ad, c2gd_b, ei_gd, z_dis, 0);
    run_gat(hd16, NDD, HDIM, hg16, NG, HDIM, w16_c2dg_s, w16_c2dg_d,
            c2dg_as, c2dg_ad, c2dg_b, ei_dg, z_gene, 0);

    // f16 copies of z for the decoder GEMMs
    cvt(z_gene, zg16, (long)NG * 128);
    cvt(z_dis,  zd16, (long)NDD * 128);

    // Decoders: gather-concat fused into WMMA GEMM, then 128->1 dot
    wmma_gemm_pair_k<<<(NL + 15) / 16, 256, 0, stream>>>(
        zg16, zd16, eli_gd, eli_gd + NL, w16_dgd, dgd_b1, Hdec, NL, 1);
    rowdot_k<<<(NL + 7) / 8, 256, 0, stream>>>(Hdec, dgd_W2, dgd_b2, s_gd, NL);

    wmma_gemm_pair_k<<<(NL + 15) / 16, 256, 0, stream>>>(
        zd16, zg16, eli_dg, eli_dg + NL, w16_ddg, ddg_b1, Hdec, NL, 1);
    rowdot_k<<<(NL + 7) / 8, 256, 0, stream>>>(Hdec, ddg_W2, ddg_b2, s_dg, NL);
}
